// Model_87866440942016
// MI455X (gfx1250) — matrix operand order repaired and checked by reading, not compiled, not run on hardware
//
#include <hip/hip_runtime.h>

// ---------------------------------------------------------------------------
// MI455X (gfx1250, wave32) implementation of the dilated-conv encoder +
// dictionary synthesis model. All GEMM-like stages run on
// v_wmma_f32_16x16x32_bf16. Activations live in bf16 (L2-resident on the
// 192MB L2). BatchNorm (training-mode stats) is folded into the LDS-staging
// affine of the consumer GEMM. The final 512ch x 2048-tap dictionary
// convolution is done as chunked WMMA GEMMs + diagonal accumulation
// (exact linear convolution, no FFT, no atomics -> deterministic).
// The packed dictionary stream in dict_conv is double-buffered in LDS via
// GLOBAL_LOAD_ASYNC_TO_LDS_B128 (ASYNCcnt) when the toolchain exposes it.
// ---------------------------------------------------------------------------

#define BSZ   2
#define TLEN  32768
#define NCH   128
#define ENC   512
#define ATOMK 2048
#define FBKS  512

typedef __attribute__((ext_vector_type(16))) __bf16 bfrag16;
typedef __attribute__((ext_vector_type(8)))  float  v8f;
typedef __attribute__((ext_vector_type(4)))  int    v4i;

#if defined(__has_builtin)
#if __has_builtin(__builtin_amdgcn_global_load_async_to_lds_b128) && \
    __has_builtin(__builtin_amdgcn_s_wait_asynccnt)
#define ASYNC_LDS 1
#endif
#endif

#ifdef ASYNC_LDS
// Builtin signature (from compiler diagnostics): first param is
// v4i in address_space(1) (global), second is the LDS-side pointer.
static __device__ __forceinline__ __attribute__((address_space(1))) v4i*
to_glb_v4(const void* p) {
    return (__attribute__((address_space(1))) v4i*)(unsigned long long)p;
}
static __device__ __forceinline__ __attribute__((address_space(3))) v4i*
to_lds_v4(void* p) {
    // generic LDS address: low 32 bits == LDS offset (ISA 10.2 aperture rules)
    return (__attribute__((address_space(3))) v4i*)(unsigned int)(unsigned long long)p;
}
#endif

union BU { bfrag16 v; unsigned short u[16]; };

static __device__ __forceinline__ unsigned short f2bf(float f) {
    union { float f; unsigned u; } x; x.f = f;
    unsigned r = x.u;
    r += 0x7fffu + ((r >> 16) & 1u);       // RNE
    return (unsigned short)(r >> 16);
}
static __device__ __forceinline__ float bf2f(unsigned short h) {
    union { unsigned u; float f; } x; x.u = ((unsigned)h) << 16;
    return x.f;
}
static __device__ __forceinline__ v8f zero8() {
    v8f z;
#pragma unroll
    for (int i = 0; i < 8; ++i) z[i] = 0.f;
    return z;
}
// 16-bit A-matrix 16x32 per-lane element -> K index (contiguous: k = 16*(lane/16) + e)
static __device__ __host__ __forceinline__ int kIndexA(int lane, int e) {
    return (lane < 16) ? e : (e + 16);
}

// ---------------------------------------------------------------------------
// Weight packing: fragment-major layout so GEMM kernels do one aligned 32B
// v16bf load per A fragment. K ordering is k = tap*CI + ci.
// ---------------------------------------------------------------------------
__global__ __launch_bounds__(256) void pack_weights(
    const float* __restrict__ src, unsigned short* __restrict__ dst,
    int M, int CI, int TAPS)
{
    int K = CI * TAPS, KC = K / 32;
    int total = (M / 16) * KC * 512;
    int idx = blockIdx.x * 256 + threadIdx.x;
    if (idx >= total) return;
    int e    = idx & 15;
    int lane = (idx >> 4) & 31;
    int kc   = (idx >> 9) % KC;
    int mt   = idx / (KC * 512);
    int m    = mt * 16 + (lane & 15);
    int k    = kc * 32 + kIndexA(lane, e);
    int tap  = k / CI, ci = k % CI;
    dst[idx] = f2bf(src[(m * CI + ci) * TAPS + tap]);
}

// dict packed per 16-tap chunk: A[k_row, c] = dict[c, k]
__global__ __launch_bounds__(256) void pack_dict(
    const float* __restrict__ src, unsigned short* __restrict__ dst)
{
    int idx = blockIdx.x * 256 + threadIdx.x;               // 1,048,576 total
    int e = idx & 15, lane = (idx >> 4) & 31;
    int kc = (idx >> 9) & 15, chunk = idx >> 13;
    int k = chunk * 16 + (lane & 15);
    int c = kc * 32 + kIndexA(lane, e);
    dst[idx] = f2bf(src[c * ATOMK + k]);
}

// ---------------------------------------------------------------------------
// Filter-bank analysis: spec[b,band,t] = sum_k fb_w[band,k] * x[b,t-256+k]
// ---------------------------------------------------------------------------
__global__ __launch_bounds__(256) void fb_kernel(
    const float* __restrict__ x, const unsigned short* __restrict__ Apk,
    unsigned short* __restrict__ spec)
{
    int t0 = blockIdx.x * 64, b = blockIdx.z;
    int tid = threadIdx.x, w = tid >> 5, lane = tid & 31;
    __shared__ __align__(32) unsigned short xw[640];
    for (int i = tid; i < 640; i += 256) {
        int t = t0 - 256 + i;
        float v = (i < 592 && t >= 0 && t < TLEN) ? x[b * TLEN + t] : 0.f;
        xw[i] = f2bf(v);
    }
    __syncthreads();

    v8f acc[4];
#pragma unroll
    for (int nt = 0; nt < 4; ++nt) acc[nt] = zero8();

    int n = lane & 15;
    int hi = (lane < 16) ? 0 : 16;
    for (int kc = 0; kc < 16; ++kc) {
        const bfrag16 a = *(const bfrag16*)(Apk + ((w * 16 + kc) * 512 + lane * 16));
        int kbase = kc * 32 + hi + n;
#pragma unroll
        for (int nt = 0; nt < 4; ++nt) {
            BU bu;
            int base = kbase + nt * 16;
#pragma unroll
            for (int e = 0; e < 16; ++e) bu.u[e] = xw[base + e];
            acc[nt] = __builtin_amdgcn_wmma_f32_16x16x32_bf16(
                false, a, false, bu.v, (short)0, acc[nt], false, false);
        }
    }
    int mhi = (lane < 16) ? 0 : 8;
#pragma unroll
    for (int nt = 0; nt < 4; ++nt)
#pragma unroll
        for (int r = 0; r < 8; ++r) {
            int band = w * 16 + r + mhi;
            int t = t0 + nt * 16 + n;
            spec[((size_t)b * NCH + band) * TLEN + t] = f2bf(acc[nt][r]);
        }
}

// ---------------------------------------------------------------------------
// Generic conv-as-GEMM over 128 input channels (Toeplitz B staged in LDS,
// channel-contiguous at pitch 160 elems = 320B so B fragments are 32B-aligned
// v16bf -> ds_load_b128 pairs). BN affine folded into staging. Padding zeros
// are NOT affine-transformed (matches reference semantics).
// ACT: 0 none, 1 leaky-relu(0.2), 2 relu. OUT_T: write time-major [b][t][co].
// SW: single overlapped window (dil==1 path, used by conv0).
// ---------------------------------------------------------------------------
template<int TAPS, int ACT, bool IN_AFF, bool RES, bool OUT_T, bool SW>
__global__ __launch_bounds__(256) void convgemm(
    const unsigned short* __restrict__ in,
    const unsigned short* __restrict__ Apk,
    const float* __restrict__ bias,
    const float* __restrict__ sc, const float* __restrict__ sh,
    const unsigned short* __restrict__ res,
    unsigned short* __restrict__ out,
    int dil, int Mtotal)
{
    constexpr int WCOLS = SW ? (64 + (TAPS - 1)) : (TAPS * 64);
    constexpr int KC = TAPS * 4;
    __shared__ __align__(32) unsigned short lds[WCOLS * 160];

    int t0 = blockIdx.x * 64, b = blockIdx.z;
    int tid = threadIdx.x, w = tid >> 5, lane = tid & 31;

    for (int idx = tid; idx < WCOLS * NCH; idx += 256) {
        int ci = idx / WCOLS;
        int c  = idx % WCOLS;
        int t;
        if (SW) { t = t0 - (TAPS / 2) + c; }
        else    { int j = c >> 6, cc = c & 63; t = t0 + (j - TAPS / 2) * dil + cc; }
        float v = 0.f;
        if (t >= 0 && t < TLEN) {
            v = bf2f(in[((size_t)b * NCH + ci) * TLEN + t]);
            if (IN_AFF) v = v * sc[ci] + sh[ci];
        }
        lds[c * 160 + ci] = f2bf(v);
    }
    __syncthreads();

    v8f acc[4];
#pragma unroll
    for (int nt = 0; nt < 4; ++nt) acc[nt] = zero8();

    int mt = blockIdx.y * 8 + w;
    int n = lane & 15;
    int hi = (lane < 16) ? 0 : 16;
    for (int j = 0; j < TAPS; ++j) {
#pragma unroll
        for (int cb = 0; cb < 4; ++cb) {
            const bfrag16 a = *(const bfrag16*)(Apk + ((mt * KC + (j * 4 + cb)) * 512 + lane * 16));
#pragma unroll
            for (int nt = 0; nt < 4; ++nt) {
                int col = (SW ? j : j * 64) + nt * 16 + n;
                const bfrag16 bf = *(const bfrag16*)(lds + col * 160 + cb * 32 + hi);
                acc[nt] = __builtin_amdgcn_wmma_f32_16x16x32_bf16(
                    false, a, false, bf, (short)0, acc[nt], false, false);
            }
        }
    }

    int mhi = (lane < 16) ? 0 : 8;
#pragma unroll
    for (int nt = 0; nt < 4; ++nt)
#pragma unroll
        for (int r = 0; r < 8; ++r) {
            int co = mt * 16 + r + mhi;
            int t = t0 + nt * 16 + n;
            float v = acc[nt][r] + bias[co];
            if (RES) {
                float hv = bf2f(res[((size_t)b * NCH + co) * TLEN + t]);
                v += hv * sc[co] + sh[co];
            }
            if (ACT == 1) v = (v >= 0.f) ? v : 0.2f * v;
            if (ACT == 2) v = fmaxf(v, 0.f);
            if (OUT_T) out[((size_t)b * TLEN + t) * Mtotal + co] = f2bf(v);
            else       out[((size_t)b * Mtotal + co) * TLEN + t] = f2bf(v);
        }
}

// ---------------------------------------------------------------------------
// BN training-mode stats over (batch,time) -> per-channel scale/shift.
// Fixed-order tree reduction (deterministic).
// ---------------------------------------------------------------------------
__global__ __launch_bounds__(256) void stats_kernel(
    const unsigned short* __restrict__ z, const float* __restrict__ g,
    const float* __restrict__ beta, float* __restrict__ sc, float* __restrict__ sh)
{
    int ch = blockIdx.x, tid = threadIdx.x;
    float s = 0.f, ss = 0.f;
    for (int b = 0; b < BSZ; ++b) {
        const unsigned short* p = z + ((size_t)b * NCH + ch) * TLEN;
        for (int t = tid; t < TLEN; t += 256) { float v = bf2f(p[t]); s += v; ss += v * v; }
    }
    __shared__ float rs[256], rq[256];
    rs[tid] = s; rq[tid] = ss; __syncthreads();
    for (int o = 128; o > 0; o >>= 1) {
        if (tid < o) { rs[tid] += rs[tid + o]; rq[tid] += rq[tid + o]; }
        __syncthreads();
    }
    if (tid == 0) {
        const float N = (float)(BSZ * TLEN);
        float m = rs[0] / N;
        float var = rq[0] / N - m * m;
        float scale = g[ch] * rsqrtf(var + 1e-5f);
        sc[ch] = scale;
        sh[ch] = beta[ch] - m * scale;
    }
}

// ---------------------------------------------------------------------------
// Dictionary synthesis: out[b,t] = sum_{c,k} dict[c,k] * enc[b,c,t-k]
// 128 chunks of 16 taps:  G[k,s] = sum_c dict[c,k]*enc[c,s]  (WMMA, K=512)
// then out[t0+i] += sum_kk G[kk, i+15-kk] via LDS diagonal sum.
// Packed dict chunks (16KB contiguous) are double-buffered in LDS with
// GLOBAL_LOAD_ASYNC_TO_LDS_B128 overlapping the current chunk's WMMAs.
// ---------------------------------------------------------------------------
__global__ __launch_bounds__(256) void dict_conv(
    const unsigned short* __restrict__ encT,
    const unsigned short* __restrict__ Dpk,
    float* __restrict__ out)
{
    int t0 = blockIdx.x * 256, b = blockIdx.z;
    int tid = threadIdx.x, w = tid >> 5, lane = tid & 31;
    __shared__ __align__(32) float G[16 * 288];
    float outAcc = 0.f;
    int n = lane & 15;
    int hi = (lane < 16) ? 0 : 16;
    int mhi = (lane < 16) ? 0 : 8;
    const unsigned short* encB = encT + (size_t)b * TLEN * ENC;

#ifdef ASYNC_LDS
    // two 16KB A-chunk buffers, filled by the async DMA path (ASYNCcnt)
    __shared__ __align__(32) unsigned short Abuf[2][8192];
    // prologue: chunk 0 -> buffer 0 (4 x b128 per thread, contiguous 16KB)
    {
        const unsigned short* gsrc = Dpk;
#pragma unroll
        for (int q = 0; q < 4; ++q) {
            int f = q * 256 + tid;          // 16-byte segment index
            __builtin_amdgcn_global_load_async_to_lds_b128(
                to_glb_v4(gsrc + f * 8), to_lds_v4(&Abuf[0][f * 8]), 0, 0);
        }
    }
#endif

    for (int chunk = 0; chunk < 128; ++chunk) {
#ifdef ASYNC_LDS
        int cur = chunk & 1;
        if (chunk + 1 < 128) {
            const unsigned short* gsrc = Dpk + (size_t)(chunk + 1) * 8192;
#pragma unroll
            for (int q = 0; q < 4; ++q) {
                int f = q * 256 + tid;
                __builtin_amdgcn_global_load_async_to_lds_b128(
                    to_glb_v4(gsrc + f * 8), to_lds_v4(&Abuf[cur ^ 1][f * 8]), 0, 0);
            }
            __builtin_amdgcn_s_wait_asynccnt(4);   // in-order: current chunk landed
        } else {
            __builtin_amdgcn_s_wait_asynccnt(0);
        }
        __syncthreads();                           // all waves' segments visible
#else
        __builtin_prefetch(Dpk + (size_t)(chunk + 1) * 8192, 0, 0);
#endif
        int sbase = t0 - chunk * 16 - 15;
        for (int j = 0; j < 3; ++j) {
            int nt = w + 8 * j;
            if (nt > 16) break;
            v8f acc = zero8();
            for (int kc = 0; kc < 16; ++kc) {
#ifdef ASYNC_LDS
                const bfrag16 a = *(const bfrag16*)(&Abuf[cur][(kc * 32 + lane) << 4]);
#else
                const bfrag16 a = *(const bfrag16*)(Dpk + (((chunk * 16 + kc) * 32 + lane) << 4));
#endif
                int s = sbase + nt * 16 + n;
                bfrag16 bf;
                if (s >= 0 && s < TLEN) {
                    bf = *(const bfrag16*)(encB + (size_t)s * ENC + kc * 32 + hi);
                } else {
                    BU z;
#pragma unroll
                    for (int e = 0; e < 16; ++e) z.u[e] = 0;
                    bf = z.v;
                }
                acc = __builtin_amdgcn_wmma_f32_16x16x32_bf16(
                    false, a, false, bf, (short)0, acc, false, false);
            }
#pragma unroll
            for (int r = 0; r < 8; ++r) G[(r + mhi) * 288 + nt * 16 + n] = acc[r];
        }
        __syncthreads();
        {
            float sum = 0.f;
#pragma unroll
            for (int kk = 0; kk < 16; ++kk) sum += G[kk * 288 + (tid + 15 - kk)];
            outAcc += sum;
        }
        __syncthreads();
    }
    out[(size_t)b * TLEN + t0 + tid] = outAcc;
}

// ---------------------------------------------------------------------------
// Host orchestration (graph-capture safe: stream launches only).
// ---------------------------------------------------------------------------
extern "C" void kernel_launch(void* const* d_in, const int* in_sizes, int n_in,
                              void* d_out, int out_size, void* d_ws, size_t ws_size,
                              hipStream_t stream)
{
    (void)in_sizes; (void)n_in; (void)out_size; (void)ws_size;
    const float* x       = (const float*)d_in[0];
    const float* fb_w    = (const float*)d_in[1];
    const float* conv0_w = (const float*)d_in[2];
    const float* conv0_b = (const float*)d_in[3];
    const float* bn0_g   = (const float*)d_in[4];
    const float* bn0_b   = (const float*)d_in[5];
    const float* dil_w   = (const float*)d_in[6];
    const float* dil_b   = (const float*)d_in[7];
    const float* pw_w    = (const float*)d_in[8];
    const float* pw_b    = (const float*)d_in[9];
    const float* bn_g    = (const float*)d_in[10];
    const float* bn_b    = (const float*)d_in[11];
    const float* up_w    = (const float*)d_in[12];
    const float* up_b    = (const float*)d_in[13];
    const float* sdict   = (const float*)d_in[14];
    float* outp = (float*)d_out;

    char* ws = (char*)d_ws;
    size_t off = 0;
    auto alloc = [&](size_t bytes) -> char* {
        char* p = ws + off;
        off = (off + bytes + 255) & ~(size_t)255;
        return p;
    };
    const size_t actB = (size_t)BSZ * NCH * TLEN * 2;        // bf16 activation
    unsigned short* fbPk  = (unsigned short*)alloc(8 * 16 * 512 * 2);
    unsigned short* c0Pk  = (unsigned short*)alloc(8 * 28 * 512 * 2);
    unsigned short* dlPk  = (unsigned short*)alloc((size_t)7 * 8 * 12 * 512 * 2);
    unsigned short* pwPk  = (unsigned short*)alloc((size_t)7 * 8 * 4 * 512 * 2);
    unsigned short* upPk  = (unsigned short*)alloc(32 * 4 * 512 * 2);
    unsigned short* dcPk  = (unsigned short*)alloc((size_t)128 * 16 * 32 * 16 * 2);
    unsigned short* spec  = (unsigned short*)alloc(actB);
    unsigned short* hA    = (unsigned short*)alloc(actB);
    unsigned short* hB    = (unsigned short*)alloc(actB);
    unsigned short* y1    = (unsigned short*)alloc(actB);
    unsigned short* encT  = (unsigned short*)alloc((size_t)BSZ * TLEN * ENC * 2);
    float* scArr = (float*)alloc(8 * NCH * 4);
    float* shArr = (float*)alloc(8 * NCH * 4);

    auto packGrid = [](int total) { return dim3((total + 255) / 256); };

    // ---- pack all weights into WMMA fragment layout (bf16) ----
    pack_weights<<<packGrid(8 * 16 * 512), 256, 0, stream>>>(fb_w, fbPk, 128, 1, FBKS);
    pack_weights<<<packGrid(8 * 28 * 512), 256, 0, stream>>>(conv0_w, c0Pk, 128, 128, 7);
    for (int i = 0; i < 7; ++i) {
        pack_weights<<<packGrid(8 * 12 * 512), 256, 0, stream>>>(
            dil_w + (size_t)i * NCH * NCH * 3, dlPk + (size_t)i * 8 * 12 * 512, 128, 128, 3);
        pack_weights<<<packGrid(8 * 4 * 512), 256, 0, stream>>>(
            pw_w + (size_t)i * NCH * NCH, pwPk + (size_t)i * 8 * 4 * 512, 128, 128, 1);
    }
    pack_weights<<<packGrid(32 * 4 * 512), 256, 0, stream>>>(up_w, upPk, 512, 128, 1);
    pack_dict<<<packGrid(128 * 16 * 32 * 16), 256, 0, stream>>>(sdict, dcPk);

    const dim3 gConv(TLEN / 64, 1, BSZ);

    // ---- filter bank ----
    fb_kernel<<<gConv, 256, 0, stream>>>(x, fbPk, spec);

    // ---- conv0 + leaky relu -> hA ; BN0 stats ----
    convgemm<7, 1, false, false, false, true><<<gConv, 256, 0, stream>>>(
        spec, c0Pk, conv0_b, nullptr, nullptr, nullptr, hA, 1, NCH);
    stats_kernel<<<NCH, 256, 0, stream>>>(hA, bn0_g, bn0_b, scArr, shArr);

    // ---- 7 dilated residual blocks ----
    const int dils[7] = {1, 3, 9, 27, 81, 243, 1};
    unsigned short* bufs[2] = {hA, hB};
    for (int i = 0; i < 7; ++i) {
        unsigned short* hin  = bufs[i & 1];
        unsigned short* hout = bufs[(i + 1) & 1];
        const float* sci = scArr + (size_t)i * NCH;
        const float* shi = shArr + (size_t)i * NCH;
        convgemm<3, 0, true, false, false, false><<<gConv, 256, 0, stream>>>(
            hin, dlPk + (size_t)i * 8 * 12 * 512, dil_b + (size_t)i * NCH,
            sci, shi, nullptr, y1, dils[i], NCH);
        convgemm<1, 1, false, true, false, false><<<gConv, 256, 0, stream>>>(
            y1, pwPk + (size_t)i * 8 * 4 * 512, pw_b + (size_t)i * NCH,
            sci, shi, hin, hout, 1, NCH);
        stats_kernel<<<NCH, 256, 0, stream>>>(
            hout, bn_g + (size_t)i * NCH, bn_b + (size_t)i * NCH,
            scArr + (size_t)(i + 1) * NCH, shArr + (size_t)(i + 1) * NCH);
    }

    // ---- up-projection + relu -> time-major encoding [b][t][512] ----
    unsigned short* h7 = bufs[7 & 1];
    convgemm<1, 2, true, false, true, false><<<dim3(TLEN / 64, 4, BSZ), 256, 0, stream>>>(
        h7, upPk, up_b, scArr + 7 * NCH, shArr + 7 * NCH, nullptr, encT, 1, ENC);

    // ---- dictionary synthesis + channel sum -> out [B,1,T] ----
    dict_conv<<<dim3(TLEN / 256, 1, BSZ), 256, 0, stream>>>(encT, dcPk, outp);
}
